// DeltaNet_31877247271546
// MI455X (gfx1250) — matrix operand order repaired and checked by reading, not compiled, not run on hardware
//
#include <hip/hip_runtime.h>
#include <hip/hip_bf16.h>

// ---------------------------------------------------------------------------
// DeltaNet forward for MI455X (gfx1250, wave32, WMMA).
// f32 -> bf16, all GEMM-shaped work on v_wmma_f32_16x16x32_bf16 (f32 acc).
// Big GEMMs use 4x2 register blocking per wave (8 WMMAs/K-step). Epilogue
// flags (bias/act/output-type) are template params so no runtime branching
// survives in the unrolled store loop. Delta-rule 256x256 state stays
// resident in LDS (bf16, 128KB of CDNA5's 320KB/WGP) across all 128 chunks.
// ---------------------------------------------------------------------------

typedef unsigned short u16;
typedef __attribute__((ext_vector_type(16))) __bf16 v16bf;
typedef __attribute__((ext_vector_type(8)))  __bf16 v8bf;
typedef __attribute__((ext_vector_type(8)))  float  v8f;

#define NB   2
#define NL   4096
#define ND   1024
#define NH   4
#define NDV  256
#define NBL  8192          // NB*NL
#define NNP  5
#define RIN_PAD  1152      // 1144 padded to 32
#define RH_PAD   2304      // 2288 padded to 32

__device__ __forceinline__ u16 f2bf(float x) {
  unsigned int u = __builtin_bit_cast(unsigned int, x);
  unsigned int r = u + 0x7FFFu + ((u >> 16) & 1u);   // RNE (NaN edge ignored)
  return (u16)(r >> 16);
}
__device__ __forceinline__ float bf2f(u16 h) {
  unsigned int u = ((unsigned int)h) << 16;
  return __builtin_bit_cast(float, u);
}
__device__ __forceinline__ v8f zero8() {
  v8f z;
#pragma unroll
  for (int i = 0; i < 8; ++i) z[i] = 0.0f;
  return z;
}

// A fragment: 16x32 (MxK). lane&15 = M row; lane>>4 selects K-half.
__device__ __forceinline__ v16bf load_a(const u16* A, int lda, int lane) {
  const u16* p = A + (lane & 15) * lda + (lane >> 4) * 16;
  return *(const v16bf*)p;
}
// B fragment: 32x16 (KxN) for C = X @ Y^T where Bt is Y (N rows, K cols).
__device__ __forceinline__ v16bf load_b(const u16* Bt, int ldb, int lane) {
  const u16* p = Bt + (lane & 15) * ldb + (lane >> 4) * 16;
  return *(const v16bf*)p;
}
// Accumulate a 16x16 tile over K (multiple of 32). Works on LDS or global.
__device__ __forceinline__ v8f mm_acc(const u16* A, int lda, const u16* Bt,
                                      int ldb, int K, int lane, v8f acc) {
  for (int kk = 0; kk < K; kk += 32) {
    v16bf a = load_a(A + kk, lda, lane);
    v16bf b = load_b(Bt + kk, ldb, lane);
    acc = __builtin_amdgcn_wmma_f32_16x16x32_bf16(false, a, false, b,
                                                  (short)0, acc, false, false);
  }
  return acc;
}

// ---------------------------------------------------------------------------
// f32 -> bf16 conversions
// ---------------------------------------------------------------------------
__global__ __launch_bounds__(256) void k_cvt(const float* s, u16* d, int n) {
  int i = blockIdx.x * 256 + threadIdx.x;
  if (i < n) d[i] = f2bf(s[i]);
}
// zero-padded strided convert (pads both rows and cols with 0)
__global__ __launch_bounds__(256) void k_cvt_pad(const float* s, u16* d,
                                                 int drows, int srows,
                                                 int scols, int dcols) {
  int i = blockIdx.x * 256 + threadIdx.x;
  if (i >= drows * dcols) return;
  int r = i / dcols, c = i - r * dcols;
  d[i] = (r < srows && c < scols) ? f2bf(s[r * scols + c]) : (u16)0;
}

// ---------------------------------------------------------------------------
// Register-tiled WMMA GEMM: C = act(A @ Bt^T + bias). Requires M%128==0,
// N%128==0. Block = 8 waves as 2x4; each wave owns 64x32 (4x2 tiles) so each
// K-step issues 8 independent WMMAs reusing 4 A-frags and 2 B-frags.
// ACT: 0 none, 1 exact GELU. CBF16: output type. HBIAS: add bias[n] (n<nbias).
// ---------------------------------------------------------------------------
template <int ACT, int CBF16, int HBIAS>
__global__ __launch_bounds__(256) void k_gemm_rt(const u16* A, const u16* Bt,
                                                 int M, int N, int K,
                                                 const float* bias, int nbias,
                                                 void* C) {
  int tid = threadIdx.x, lane = tid & 31, wv = tid >> 5;
  int tm = blockIdx.y * 128 + (wv >> 2) * 64;   // 4 M-tiles from here
  int tn = blockIdx.x * 128 + (wv & 3) * 32;    // 2 N-tiles from here
  const u16* Ab0 = A + (long)tm * K;
  const u16* Bb0 = Bt + (long)tn * K;
  v8f acc[8];
#pragma unroll
  for (int t = 0; t < 8; ++t) acc[t] = zero8();

  for (int kk = 0; kk < K; kk += 32) {
    __builtin_prefetch(Ab0 + kk + 256, 0, 1);           // global_prefetch_b8
    __builtin_prefetch(Bb0 + kk + 256, 0, 1);
    v16bf af[4], bf_[2];
#pragma unroll
    for (int i = 0; i < 4; ++i)
      af[i] = load_a(Ab0 + (long)(16 * i) * K + kk, K, lane);
#pragma unroll
    for (int j = 0; j < 2; ++j)
      bf_[j] = load_b(Bb0 + (long)(16 * j) * K + kk, K, lane);
#pragma unroll
    for (int i = 0; i < 4; ++i)
#pragma unroll
      for (int j = 0; j < 2; ++j)
        acc[i * 2 + j] = __builtin_amdgcn_wmma_f32_16x16x32_bf16(
            false, af[i], false, bf_[j], (short)0, acc[i * 2 + j],
            false, false);
  }

  int nl = lane & 15, hi = lane >> 4;
#pragma unroll
  for (int j = 0; j < 2; ++j) {
    int n = tn + j * 16 + nl;
    float bv = 0.0f;
    if (HBIAS) bv = (n < nbias) ? bias[n] : 0.0f;
#pragma unroll
    for (int i = 0; i < 4; ++i) {
#pragma unroll
      for (int r = 0; r < 8; ++r) {
        int m = tm + i * 16 + r + 8 * hi;
        float v = acc[i * 2 + j][r] + bv;
        if (ACT == 1) v = 0.5f * v * (1.0f + erff(v * 0.70710678118f));
        long idx = (long)m * N + n;
        if (CBF16) ((u16*)C)[idx] = f2bf(v);
        else       ((float*)C)[idx] = v;
      }
    }
  }
}

// ---------------------------------------------------------------------------
// Simple WMMA GEMM (one 16x16 tile per wave) for small N (logits, N=32).
// Block = 8 waves as 4x2 -> 64x32 per block. f32 output, no bias/act.
// ---------------------------------------------------------------------------
__global__ __launch_bounds__(256) void k_gemm_s(const u16* A, const u16* Bt,
                                                int M, int N, int K,
                                                float* C) {
  int tid = threadIdx.x, lane = tid & 31, wv = tid >> 5;
  int tm = blockIdx.y * 64 + (wv >> 1) * 16;
  int tn = blockIdx.x * 32 + (wv & 1) * 16;
  v8f acc = zero8();
  const u16* Ab = A + (long)tm * K;
  const u16* Bb = Bt + (long)tn * K;
  for (int kk = 0; kk < K; kk += 32) {
    __builtin_prefetch(Ab + kk + 128, 0, 1);
    v16bf a = load_a(Ab + kk, K, lane);
    v16bf b = load_b(Bb + kk, K, lane);
    acc = __builtin_amdgcn_wmma_f32_16x16x32_bf16(false, a, false, b,
                                                  (short)0, acc, false, false);
  }
  int nl = lane & 15, hi = lane >> 4;
  int n = tn + nl;
#pragma unroll
  for (int r = 0; r < 8; ++r) {
    int m = tm + r + 8 * hi;
    C[(long)m * N + n] = acc[r];
  }
}

// ---------------------------------------------------------------------------
// Depthwise causal FIR conv (+optional SiLU). x,y bf16 (B,L,D); w f32 (D,Kc).
// ---------------------------------------------------------------------------
__global__ __launch_bounds__(256) void k_conv(const u16* x, const float* w,
                                              u16* y, int Kc, int silu) {
  int idx = blockIdx.x * 256 + threadIdx.x;
  if (idx >= NBL * ND) return;
  int d = idx & (ND - 1);
  int bl = idx >> 10;
  int b = bl >> 12, l = bl & (NL - 1);
  float s = 0.0f;
  for (int t = 0; t < Kc; ++t) {
    int li = l - (Kc - 1) + t;
    if (li >= 0) s += bf2f(x[((b << 12) + li) * ND + d]) * w[d * Kc + t];
  }
  if (silu) s = s / (1.0f + expf(-s));
  y[idx] = f2bf(s);
}

// ---------------------------------------------------------------------------
// beta = sigmoid(hidden @ Wb^T). One wave per row, wave32 shuffle reduce.
// ---------------------------------------------------------------------------
__global__ __launch_bounds__(256) void k_beta(const float* hid, const float* Wb,
                                              float* beta) {
  int lane = threadIdx.x & 31, wv = threadIdx.x >> 5;
  int g = blockIdx.x * 8 + wv;
  for (int h = 0; h < NH; ++h) {
    float s = 0.0f;
    for (int i = lane; i < ND; i += 32) s += hid[g * ND + i] * Wb[h * ND + i];
#pragma unroll
    for (int o = 16; o > 0; o >>= 1) s += __shfl_xor(s, o, 32);
    if (lane == 0) beta[g * NH + h] = 1.0f / (1.0f + expf(-s));
  }
}

// ---------------------------------------------------------------------------
// Chunkwise delta rule. One block per (b,h); S^T (dv x dk) resident in LDS
// as bf16 for all 128 chunks. All matmuls are 16x16x32 WMMA on LDS tiles.
// ---------------------------------------------------------------------------
__global__ __launch_bounds__(256) void k_delta(const u16* qg, const u16* kg,
                                               const u16* vg, const float* beta,
                                               u16* dout) {
  extern __shared__ char dsm[];
  u16* St  = (u16*)dsm;          // [256][256] S^T  (dv-major)
  u16* QS  = St  + 65536;        // [32][256] q normed
  u16* KS  = QS  + 8192;         // [32][256] k normed
  u16* KBS = KS  + 8192;         // [32][256] k*beta
  u16* WW  = KBS + 8192;         // [32][256] w = attn@kb
  u16* KT  = WW  + 8192;         // [256][32] k^T
  u16* VT  = KT  + 8192;         // [256][32] (v*beta)^T
  u16* KBT = VT  + 8192;         // [256][32] (k*beta)^T
  u16* UT  = KBT + 8192;         // [256][32] u^T then u_i^T
  float* ATT = (float*)(UT + 8192);  // [32][32]
  u16*   ATB = (u16*)(ATT + 1024);   // [32][32] bf16
  float* LAf = (float*)(ATB + 1024); // [32][32]
  u16*   LAB = (u16*)(LAf + 1024);   // [32][32] bf16
  float* QSS = (float*)(LAB + 1024); // [32]
  float* KSS = QSS + 32;             // [32]

  int tid = threadIdx.x, lane = tid & 31, wv = tid >> 5;
  int bb = blockIdx.x >> 2, h = blockIdx.x & 3;

  for (int i = tid; i < 65536; i += 256) St[i] = 0;
  __syncthreads();

  for (int c = 0; c < NL / 32; ++c) {
    // ---- stage 1: load chunk, l2norm q/k, apply beta; build transposes ----
    int r = tid >> 3, p8 = tid & 7;
    int l = c * 32 + r;
    int grow = bb * NL + l;
    const u16* qrow = qg + grow * ND + h * NDV;
    const u16* krow = kg + grow * ND + h * NDV;
    const u16* vrow = vg + grow * ND + h * NDV;
    if (tid < 32) { QSS[tid] = 0.0f; KSS[tid] = 0.0f; }
    __syncthreads();
    float sq = 0.0f, sk = 0.0f;
    for (int e = p8 * 32; e < p8 * 32 + 32; ++e) {
      float fq = bf2f(qrow[e]); sq += fq * fq;
      float fk = bf2f(krow[e]); sk += fk * fk;
    }
    atomicAdd(&QSS[r], sq);                    // ds_add_f32
    atomicAdd(&KSS[r], sk);
    __syncthreads();
    float rq = rsqrtf(QSS[r] + 1e-12f);
    float rk = rsqrtf(KSS[r] + 1e-12f);
    float bet = beta[grow * NH + h];
    for (int e = p8 * 32; e < p8 * 32 + 32; ++e) {
      float qn = bf2f(qrow[e]) * rq;
      float kn = bf2f(krow[e]) * rk;
      float kb = kn * bet;
      float vv = bf2f(vrow[e]) * bet;
      QS[r * 256 + e]  = f2bf(qn);
      u16 kh = f2bf(kn);  KS[r * 256 + e] = kh;  KT[e * 32 + r]  = kh;
      u16 bh = f2bf(kb);  KBS[r * 256 + e] = bh; KBT[e * 32 + r] = bh;
      VT[e * 32 + r] = f2bf(vv);
    }
    __syncthreads();

    // ---- stage 2: attn = strict_lower(-(kb @ k^T)), 4 WMMA tiles ----
    if (wv < 4) {
      int ti = wv >> 1, tj = wv & 1;
      v8f acc = mm_acc(KBS + ti * 16 * 256, 256, KS + tj * 16 * 256, 256,
                       256, lane, zero8());
      int nl = lane & 15, hi = lane >> 4;
#pragma unroll
      for (int r8 = 0; r8 < 8; ++r8) {
        int mg = ti * 16 + r8 + 8 * hi, ng = tj * 16 + nl;
        ATT[mg * 32 + ng] = (mg > ng) ? -acc[r8] : 0.0f;
      }
    }
    __syncthreads();

    // ---- stage 3: triangular inverse recurrence (wave 0), then +I ----
    if (wv == 0) {
      for (int i = 1; i < 32; ++i) {
        float s = 0.0f;
        for (int kx = 0; kx < 32; ++kx)
          s += ATT[i * 32 + kx] * ATT[kx * 32 + lane];
        ATT[i * 32 + lane] += s;
      }
      ATT[lane * 32 + lane] += 1.0f;
    }
    __syncthreads();
    for (int i = tid * 4; i < tid * 4 + 4; ++i) ATB[i] = f2bf(ATT[i]);
    __syncthreads();

    // ---- stage 4: u = attn@v (store u^T), w = attn@kb; 64 tiles, K=32 ----
    for (int job = wv; job < 64; job += 8) {
      int isw = job >= 32, t = job & 31, tm = t >> 4, tn = t & 15;
      const u16* Bs = isw ? (KBT + tn * 16 * 32) : (VT + tn * 16 * 32);
      v8f acc = mm_acc(ATB + tm * 16 * 32, 32, Bs, 32, 32, lane, zero8());
      int nl = lane & 15, hi = lane >> 4;
#pragma unroll
      for (int r8 = 0; r8 < 8; ++r8) {
        int m = tm * 16 + r8 + 8 * hi, n = tn * 16 + nl;
        if (isw) WW[m * 256 + n] = f2bf(acc[r8]);
        else     UT[n * 32 + m]  = f2bf(acc[r8]);
      }
    }
    __syncthreads();

    // ---- stage 5: u_i = u - w@S (B-lanes = rows of S^T), la = tril(q@k^T) --
    for (int job = wv; job < 36; job += 8) {
      if (job < 32) {
        int tm = job >> 4, tn = job & 15;
        v8f acc = mm_acc(WW + tm * 16 * 256, 256, St + tn * 16 * 256, 256,
                         256, lane, zero8());
        int nl = lane & 15, hi = lane >> 4;
#pragma unroll
        for (int r8 = 0; r8 < 8; ++r8) {
          int m = tm * 16 + r8 + 8 * hi, n = tn * 16 + nl;
          float u = bf2f(UT[n * 32 + m]);
          UT[n * 32 + m] = f2bf(u - acc[r8]);
        }
      } else {
        int t = job - 32, ti = t >> 1, tj = t & 1;
        v8f acc = mm_acc(QS + ti * 16 * 256, 256, KS + tj * 16 * 256, 256,
                         256, lane, zero8());
        int nl = lane & 15, hi = lane >> 4;
#pragma unroll
        for (int r8 = 0; r8 < 8; ++r8) {
          int mg = ti * 16 + r8 + 8 * hi, ng = tj * 16 + nl;
          LAf[mg * 32 + ng] = (mg >= ng) ? acc[r8] : 0.0f;
        }
      }
    }
    __syncthreads();
    for (int i = tid * 4; i < tid * 4 + 4; ++i) LAB[i] = f2bf(LAf[i]);
    __syncthreads();

    // ---- stage 6: o = q@S + la@u_i -> global delta branch ----
    for (int job = wv; job < 32; job += 8) {
      int tm = job >> 4, tn = job & 15;
      v8f acc = mm_acc(QS + tm * 16 * 256, 256, St + tn * 16 * 256, 256,
                       256, lane, zero8());
      acc = mm_acc(LAB + tm * 16 * 32, 32, UT + tn * 16 * 32, 32,
                   32, lane, acc);
      int nl = lane & 15, hi = lane >> 4;
#pragma unroll
      for (int r8 = 0; r8 < 8; ++r8) {
        int mg = tm * 16 + r8 + 8 * hi, ng = tn * 16 + nl;
        int l2 = c * 32 + mg;
        dout[(bb * NL + l2) * ND + h * NDV + ng] = f2bf(acc[r8]);
      }
    }
    __syncthreads();   // o must read old S before the update below

    // ---- stage 7: S^T += (k^T @ u_i)^T; 256 tiles, K=32 ----
    for (int job = wv; job < 256; job += 8) {
      int tm = job >> 4, tn = job & 15;      // tm: dk tile, tn: dv tile
      int nl = lane & 15, hi = lane >> 4;
      v8f cin;
#pragma unroll
      for (int r8 = 0; r8 < 8; ++r8)
        cin[r8] = bf2f(St[(tn * 16 + nl) * 256 + tm * 16 + r8 + 8 * hi]);
      v8f acc = mm_acc(KT + tm * 16 * 32, 32, UT + tn * 16 * 32, 32,
                       32, lane, cin);
#pragma unroll
      for (int r8 = 0; r8 < 8; ++r8)
        St[(tn * 16 + nl) * 256 + tm * 16 + r8 + 8 * hi] = f2bf(acc[r8]);
    }
    __syncthreads();
  }
}

// ---------------------------------------------------------------------------
// Router features: hidden copy + per-(branch,head) stats + cross-corrs into
// zero-padded router_in (stride 1152). One block per (b,l).
// ---------------------------------------------------------------------------
__global__ __launch_bounds__(256) void k_feats(const float* hid,
                                               const u16* b0, const u16* b1,
                                               const u16* b2, const u16* b3,
                                               const u16* b4, u16* rin) {
  __shared__ u16 ls[5 * ND];
  int tid = threadIdx.x, g = blockIdx.x;
  const u16* brs[5] = {b0, b1, b2, b3, b4};
  for (int i = tid; i < 5 * ND; i += 256) {
    int p = i >> 10, j = i & (ND - 1);
    ls[i] = brs[p][g * ND + j];
  }
  u16* rb = rin + g * RIN_PAD;
  for (int i = tid; i < ND; i += 256) rb[i] = f2bf(hid[g * ND + i]);
  if (tid < 8) rb[1144 + tid] = 0;            // pad cols to 1152
  __syncthreads();

  if (tid < 20) {                              // stats: mean,var,max,entropy
    int p = tid >> 2, h = tid & 3;
    const u16* x = ls + p * ND + h * NDV;
    float s = 0.0f, ss = 0.0f, mx = -3.0e38f;
    for (int e = 0; e < NDV; ++e) {
      float v = bf2f(x[e]);
      s += v; ss += v * v; mx = fmaxf(mx, v);
    }
    float mean = s * (1.0f / NDV);
    float var = (ss - NDV * mean * mean) * (1.0f / (NDV - 1));
    float se = 0.0f;
    for (int e = 0; e < NDV; ++e) se += expf(bf2f(x[e]) - mx);
    float ent = 0.0f;
    for (int e = 0; e < NDV; ++e) {
      float sm = expf(bf2f(x[e]) - mx) / se;
      ent -= sm * logf(sm + 1e-8f);
    }
    int base = ND + p * 16;
    rb[base + 0 * 4 + h] = f2bf(mean);
    rb[base + 1 * 4 + h] = f2bf(var);
    rb[base + 2 * 4 + h] = f2bf(mx);
    rb[base + 3 * 4 + h] = f2bf(ent);
  }
  if (tid >= 64 && tid < 104) {                // 10 cross-corrs x 4 heads
    const int PI[10] = {0, 0, 0, 0, 1, 1, 1, 2, 2, 3};
    const int PJ[10] = {1, 2, 3, 4, 2, 3, 4, 3, 4, 4};
    int idx = tid - 64, pr = idx >> 2, h = idx & 3;
    const u16* xi = ls + PI[pr] * ND + h * NDV;
    const u16* xj = ls + PJ[pr] * ND + h * NDV;
    float sij = 0.0f, sii = 0.0f, sjj = 0.0f;
    for (int e = 0; e < NDV; ++e) {
      float a = bf2f(xi[e]), b = bf2f(xj[e]);
      sij += a * b; sii += a * a; sjj += b * b;
    }
    float den = sqrtf(sii) * sqrtf(sjj) + 1e-8f;
    rb[ND + 80 + pr * 4 + h] = f2bf(sij / den);
  }
}

// ---------------------------------------------------------------------------
// Gate + mix + RMS norm. One wave per (b,l,h).
// ---------------------------------------------------------------------------
__global__ __launch_bounds__(256) void k_mix(const float* logits,
                                             const float* b2v,
                                             const u16* b0, const u16* b1,
                                             const u16* b2b, const u16* b3,
                                             const u16* b4, const float* nw,
                                             u16* mixed) {
  int lane = threadIdx.x & 31, wv = threadIdx.x >> 5;
  int unit = blockIdx.x * 8 + wv;
  int g = unit >> 2, h = unit & 3;
  const u16* brs[5] = {b0, b1, b2b, b3, b4};
  float lg[NNP];
  float mx = -3.0e38f;
#pragma unroll
  for (int p = 0; p < NNP; ++p) {
    lg[p] = logits[g * 32 + h * NNP + p] + b2v[h * NNP + p];
    mx = fmaxf(mx, lg[p]);
  }
  float se = 0.0f;
#pragma unroll
  for (int p = 0; p < NNP; ++p) { lg[p] = expf(lg[p] - mx); se += lg[p]; }
  float pr[NNP];
#pragma unroll
  for (int p = 0; p < NNP; ++p) pr[p] = (lg[p] / se) * 0.95f + 0.01f;

  float o[8], ssq = 0.0f;
#pragma unroll
  for (int e = 0; e < 8; ++e) {
    int d = e * 32 + lane;
    float acc = 0.0f;
#pragma unroll
    for (int p = 0; p < NNP; ++p)
      acc += pr[p] * bf2f(brs[p][g * ND + h * NDV + d]);
    o[e] = acc; ssq += acc * acc;
  }
#pragma unroll
  for (int off = 16; off > 0; off >>= 1) ssq += __shfl_xor(ssq, off, 32);
  float scale = rsqrtf(ssq * (1.0f / NDV) + 1e-5f);
#pragma unroll
  for (int e = 0; e < 8; ++e) {
    int d = e * 32 + lane;
    mixed[g * ND + h * NDV + d] = f2bf(o[e] * scale * nw[d]);
  }
}

// ---------------------------------------------------------------------------
extern "C" void kernel_launch(void* const* d_in, const int* in_sizes, int n_in,
                              void* d_out, int out_size, void* d_ws,
                              size_t ws_size, hipStream_t stream) {
  (void)in_sizes; (void)n_in; (void)out_size; (void)ws_size;
  const float* hid   = (const float*)d_in[0];
  const float* Wq    = (const float*)d_in[1];
  const float* Wk    = (const float*)d_in[2];
  const float* Wv    = (const float*)d_in[3];
  const float* wqc   = (const float*)d_in[4];
  const float* wkc   = (const float*)d_in[5];
  const float* wvc   = (const float*)d_in[6];
  const float* Wb    = (const float*)d_in[7];
  const float* firS  = (const float*)d_in[8];
  const float* firM  = (const float*)d_in[9];
  const float* firL  = (const float*)d_in[10];
  const float* firI  = (const float*)d_in[11];
  const float* W1    = (const float*)d_in[12];
  const float* b1    = (const float*)d_in[13];
  const float* W2    = (const float*)d_in[14];
  const float* b2    = (const float*)d_in[15];
  const float* nw    = (const float*)d_in[16];
  const float* Wo    = (const float*)d_in[17];

  size_t off = 0;
  auto alloc = [&](size_t bytes) {
    void* p = (char*)d_ws + off;
    off += (bytes + 255) & ~(size_t)255;
    return p;
  };
  u16* hsb  = (u16*)alloc((size_t)NBL * ND * 2);
  u16* Wqb  = (u16*)alloc((size_t)ND * ND * 2);
  u16* Wkb  = (u16*)alloc((size_t)ND * ND * 2);
  u16* Wvb  = (u16*)alloc((size_t)ND * ND * 2);
  u16* Wob  = (u16*)alloc((size_t)ND * ND * 2);
  u16* W1p  = (u16*)alloc((size_t)RH_PAD * RIN_PAD * 2);
  u16* W2p  = (u16*)alloc((size_t)32 * RH_PAD * 2);
  u16* qpre = (u16*)alloc((size_t)NBL * ND * 2);
  u16* kpre = (u16*)alloc((size_t)NBL * ND * 2);
  u16* vpre = (u16*)alloc((size_t)NBL * ND * 2);
  u16* qc   = (u16*)alloc((size_t)NBL * ND * 2);
  u16* kc   = (u16*)alloc((size_t)NBL * ND * 2);
  u16* vc   = (u16*)alloc((size_t)NBL * ND * 2);
  float* bet = (float*)alloc((size_t)NBL * NH * 4);
  u16* br0  = (u16*)alloc((size_t)NBL * ND * 2);
  u16* br1  = (u16*)alloc((size_t)NBL * ND * 2);
  u16* br2  = (u16*)alloc((size_t)NBL * ND * 2);
  u16* br3  = (u16*)alloc((size_t)NBL * ND * 2);  // delta branch
  u16* br4  = (u16*)alloc((size_t)NBL * ND * 2);
  u16* rin  = (u16*)alloc((size_t)NBL * RIN_PAD * 2);
  u16* hrt  = (u16*)alloc((size_t)NBL * RH_PAD * 2);
  float* lgt = (float*)alloc((size_t)NBL * 32 * 4);
  u16* mixd = (u16*)alloc((size_t)NBL * ND * 2);

  const int hn = NBL * ND, wn = ND * ND;
  k_cvt<<<(hn + 255) / 256, 256, 0, stream>>>(hid, hsb, hn);
  k_cvt<<<(wn + 255) / 256, 256, 0, stream>>>(Wq, Wqb, wn);
  k_cvt<<<(wn + 255) / 256, 256, 0, stream>>>(Wk, Wkb, wn);
  k_cvt<<<(wn + 255) / 256, 256, 0, stream>>>(Wv, Wvb, wn);
  k_cvt<<<(wn + 255) / 256, 256, 0, stream>>>(Wo, Wob, wn);
  {
    int t1 = RH_PAD * RIN_PAD;
    k_cvt_pad<<<(t1 + 255) / 256, 256, 0, stream>>>(W1, W1p, RH_PAD, 2288,
                                                    1144, RIN_PAD);
    int t2 = 32 * RH_PAD;
    k_cvt_pad<<<(t2 + 255) / 256, 256, 0, stream>>>(W2, W2p, 32, 20,
                                                    2288, RH_PAD);
  }

  // Projections: register-tiled WMMA GEMM, block tile 128x128, bf16 out
  dim3 gP(ND / 128, NBL / 128);
  k_gemm_rt<0, 1, 0><<<gP, 256, 0, stream>>>(hsb, Wqb, NBL, ND, ND,
                                             nullptr, 0, qpre);
  k_gemm_rt<0, 1, 0><<<gP, 256, 0, stream>>>(hsb, Wkb, NBL, ND, ND,
                                             nullptr, 0, kpre);
  k_gemm_rt<0, 1, 0><<<gP, 256, 0, stream>>>(hsb, Wvb, NBL, ND, ND,
                                             nullptr, 0, vpre);

  int cg = (NBL * ND) / 256;
  k_conv<<<cg, 256, 0, stream>>>(qpre, wqc, qc, 4, 1);
  k_conv<<<cg, 256, 0, stream>>>(kpre, wkc, kc, 4, 1);
  k_conv<<<cg, 256, 0, stream>>>(vpre, wvc, vc, 4, 1);
  k_beta<<<NBL / 8, 256, 0, stream>>>(hid, Wb, bet);

  // FIR branches on post-SiLU v
  k_conv<<<cg, 256, 0, stream>>>(vc, firS, br0, 3, 0);
  k_conv<<<cg, 256, 0, stream>>>(vc, firM, br1, 7, 0);
  k_conv<<<cg, 256, 0, stream>>>(vc, firL, br2, 25, 0);
  k_conv<<<cg, 256, 0, stream>>>(vc, firI, br4, 1, 0);

  // delta rule: 8 blocks (b,h), 274688 B dynamic LDS (< 320KB WGP limit)
  k_delta<<<NB * NH, 256, 274688, stream>>>(qc, kc, vc, bet, br3);

  k_feats<<<NBL, 256, 0, stream>>>(hid, br0, br1, br2, br3, br4, rin);

  // Router layer 1 (bias + exact GELU): 8192 x 2304 x 1152, bf16 out
  dim3 gR1(RH_PAD / 128, NBL / 128);
  k_gemm_rt<1, 1, 1><<<gR1, 256, 0, stream>>>(rin, W1p, NBL, RH_PAD, RIN_PAD,
                                              b1, 2288, hrt);
  // Router layer 2 (logits): N=32, simple kernel, f32 out
  dim3 gR2(1, NBL / 64);
  k_gemm_s<<<gR2, 256, 0, stream>>>(hrt, W2p, NBL, 32, RH_PAD, lgt);

  k_mix<<<(NBL * NH) / 8, 256, 0, stream>>>(lgt, b2, br0, br1, br2, br3, br4,
                                            nw, mixd);

  // Output projection -> f32 d_out
  k_gemm_rt<0, 0, 0><<<gP, 256, 0, stream>>>(mixd, Wob, NBL, ND, ND,
                                             nullptr, 0, d_out);
}
